// LNLSTMLayer_35510789603633
// MI455X (gfx1250) — compile-verified, matrix operand order checked by reading, not run on hardware
//
#include <hip/hip_runtime.h>
#include <stdint.h>

// LN-LSTM for MI455X (gfx1250, wave32, WMMA bf16 16x16x32).
// Phase 1: gx = LN(x @ Wx^T) as a fused WMMA GEMM + LN (throughput, whole GPU).
// Phase 2: persistent scan kernel; 4 WGs, each owns 16 batch rows, h/c in LDS.

#define B_ 64
#define T_ 512
#define E_ 256
#define H_ 512
#define G_ 2048
#define EPS_ 1e-5f

typedef __attribute__((ext_vector_type(16))) __bf16 v16bf;
typedef __attribute__((ext_vector_type(8)))  float  v8f;
typedef __attribute__((ext_vector_type(4)))  unsigned int u32x4;

union Frag16 { u32x4 u[2]; v16bf v; };

__device__ __forceinline__ uint16_t f2bf(float f) {
  union { float f; uint32_t u; } c; c.f = f;
  uint32_t u = c.u;
  return (uint16_t)((u + 0x7FFFu + ((u >> 16) & 1u)) >> 16);   // RNE
}
__device__ __forceinline__ float sig_(float x)  { return 1.0f / (1.0f + __expf(-x)); }
__device__ __forceinline__ float tanh_(float x) { return 1.0f - 2.0f / (__expf(2.0f * x) + 1.0f); }
__device__ __forceinline__ float hsum16(float v) {   // reduce across 16-lane half (rows stay put)
  v += __shfl_xor(v, 1, 32);
  v += __shfl_xor(v, 2, 32);
  v += __shfl_xor(v, 4, 32);
  v += __shfl_xor(v, 8, 32);
  return v;
}

// ---------------- Phase 0: conversions / fragment packing ----------------

__global__ void k_cvt_bf16(const float* __restrict__ src, uint16_t* __restrict__ dst, int n) {
  int i = blockIdx.x * blockDim.x + threadIdx.x;
  if (i < n) dst[i] = f2bf(src[i]);
}

// Pack W [2048, K] (row-major, g-major) into WMMA B-fragment order:
// Wp[kc][nt][lane][16 halfs], so each lane of a wave reads its whole 32B fragment
// contiguously (coalesced 1KB per wave per fragment).
__global__ void k_pack_w(const float* __restrict__ W, uint16_t* __restrict__ Wp,
                         int K, int total) {
  int tid = blockIdx.x * blockDim.x + threadIdx.x;
  if (tid >= total) return;
  int lane = tid & 31;
  int nt   = (tid >> 5) & 127;
  int kc   = tid >> 12;           // 128*32 = 4096 per kc
  int n  = nt * 16 + (lane & 15);
  int hi = lane >> 4;
  alignas(16) uint16_t tmp[16];
#pragma unroll
  for (int j = 0; j < 8; ++j)
#pragma unroll
    for (int p = 0; p < 2; ++p) {
      int k = kc * 32 + (j >> 2) * 16 + hi * 8 + (j & 3) * 2 + p;
      tmp[j * 2 + p] = f2bf(W[(size_t)n * K + k]);
    }
  u32x4* d = (u32x4*)(Wp + (size_t)tid * 16);
  const u32x4* s = (const u32x4*)tmp;
  d[0] = s[0]; d[1] = s[1];
}

// ---------------- Phase 1: gx = LN(x @ Wx^T) ----------------
// One WG (8 waves) per 16 rows; WG covers the full N=2048, so LN is in-workgroup.
// Each wave: 16 N-tiles, K=256 in 8 chunks -> 128 WMMAs, 16 f32 accumulators.

__global__ __launch_bounds__(256) void k_gx(const uint16_t* __restrict__ xb,
                                            const uint16_t* __restrict__ wxp,
                                            const float* __restrict__ lng,
                                            const float* __restrict__ lnb,
                                            float* __restrict__ gx) {
  __shared__ alignas(16) uint16_t xs[16 * 256];   // 8 KB A tile (bf16)
  __shared__ float gs[16], gq[16];

  const int tid  = threadIdx.x;
  const int wave = tid >> 5, lane = tid & 31, hi = lane >> 4, lm = lane & 15;
  const int mbase = blockIdx.x * 16;

  { // stage x tile (coalesced 32B/thread)
    const u32x4* src = (const u32x4*)(xb + (size_t)mbase * E_);
    u32x4* dst = (u32x4*)xs;
    dst[tid * 2 + 0] = src[tid * 2 + 0];
    dst[tid * 2 + 1] = src[tid * 2 + 1];
  }
  if (tid < 16) { gs[tid] = 0.f; gq[tid] = 0.f; }
  __syncthreads();

  v8f acc[16] = {};
#pragma unroll
  for (int kc = 0; kc < 8; ++kc) {
    Frag16 a;
    a.u[0] = *(const u32x4*)(xs + lm * 256 + kc * 32 + hi * 8);
    a.u[1] = *(const u32x4*)(xs + lm * 256 + kc * 32 + 16 + hi * 8);
#pragma unroll
    for (int j = 0; j < 16; ++j) {
      int nt = j * 8 + wave;
      const u32x4* bp = (const u32x4*)(wxp + ((size_t)(kc * 128 + nt) * 32 + lane) * 16);
      Frag16 b; b.u[0] = bp[0]; b.u[1] = bp[1];
      acc[j] = __builtin_amdgcn_wmma_f32_16x16x32_bf16(false, a.v, false, b.v,
                                                       (short)0, acc[j], false, false);
    }
  }

  // per-row mean/var over all 2048 columns
  float ps[8] = {}, pq[8] = {};
#pragma unroll
  for (int j = 0; j < 16; ++j)
#pragma unroll
    for (int v = 0; v < 8; ++v) { float p = acc[j][v]; ps[v] += p; pq[v] += p * p; }
#pragma unroll
  for (int v = 0; v < 8; ++v) {
    float s = hsum16(ps[v]);
    float q = hsum16(pq[v]);
    if (lm == 0) { atomicAdd(&gs[v + 8 * hi], s); atomicAdd(&gq[v + 8 * hi], q); }
  }
  __syncthreads();

  float mu[8], rs[8];
#pragma unroll
  for (int v = 0; v < 8; ++v) {
    int m = v + 8 * hi;
    float m1 = gs[m] * (1.0f / G_);
    float m2 = gq[m] * (1.0f / G_);
    mu[v] = m1;
    rs[v] = rsqrtf(fmaxf(m2 - m1 * m1, 0.f) + EPS_);
  }
#pragma unroll
  for (int j = 0; j < 16; ++j) {
    int col = (j * 8 + wave) * 16 + lm;
    float g = lng[col], b = lnb[col];
#pragma unroll
    for (int v = 0; v < 8; ++v) {
      int m = v + 8 * hi;
      gx[(size_t)(mbase + m) * G_ + col] = (acc[j][v] - mu[v]) * rs[v] * g + b;
    }
  }
}

// ---------------- Phase 2: persistent LN-LSTM scan ----------------
// 4 WGs x 1024 threads. WG owns batch rows [wg*16, wg*16+16); loops t=0..511.
// Wave w owns H-columns [w*16, w*16+16); its 4 accumulators are the f,i,o,g
// tiles at the SAME H-columns, so gate math needs no cross-wave exchange.

__global__ __launch_bounds__(1024) void k_scan(const float* __restrict__ gx,
                                               const uint16_t* __restrict__ whp,
                                               const float* __restrict__ bias,
                                               const float* __restrict__ lnhg,
                                               const float* __restrict__ lnhb,
                                               const float* __restrict__ lncg,
                                               const float* __restrict__ lncb,
                                               const int* __restrict__ lengths,
                                               float* __restrict__ seq,
                                               float* __restrict__ lasth) {
  __shared__ alignas(16) uint16_t hbf[16 * 512];   // 16 KB  h (bf16, WMMA A source)
  __shared__ float hf[16 * 512];                   // 32 KB  h (f32 state)
  __shared__ float cf[16 * 512];                   // 32 KB  c (f32 state)
  __shared__ float gs[16], gq[16], cs[16], cq[16]; // LN reduction scratch

  const int tid  = threadIdx.x;
  const int wave = tid >> 5, lane = tid & 31, hi = lane >> 4, lm = lane & 15;
  const int rb = blockIdx.x * 16;
  const int nh = wave * 16 + lm;                   // this lane's H column

  for (int i = tid; i < 16 * 512; i += 1024) { hbf[i] = 0; hf[i] = 0.f; cf[i] = 0.f; }
  if (tid < 16) { gs[tid] = 0.f; gq[tid] = 0.f; cs[tid] = 0.f; cq[tid] = 0.f; }

  float lgh[4], lhb_[4], bs[4];
#pragma unroll
  for (int e = 0; e < 4; ++e) {
    int col = e * 512 + nh;
    lgh[e] = lnhg[col]; lhb_[e] = lnhb[col]; bs[e] = bias[col];
  }
  const float lcg = lncg[nh], lcb = lncb[nh];
  int len8[8];
#pragma unroll
  for (int v = 0; v < 8; ++v) len8[v] = lengths[rb + v + 8 * hi];
  __syncthreads();

  for (int t = 0; t < T_; ++t) {
    // prefetch gx for this step (hides HBM latency under the GEMM)
    float gxv[4][8];
#pragma unroll
    for (int e = 0; e < 4; ++e)
#pragma unroll
      for (int v = 0; v < 8; ++v) {
        int brow = rb + v + 8 * hi;
        gxv[e][v] = gx[((size_t)brow * T_ + t) * G_ + e * 512 + nh];
      }

    // P = h @ Wh^T : per wave, 4 gate tiles at same H-cols, K=512 in 16 chunks
    v8f acc[4] = {};
#pragma unroll
    for (int kc = 0; kc < 16; ++kc) {
      Frag16 a;
      a.u[0] = *(const u32x4*)(hbf + lm * 512 + kc * 32 + hi * 8);
      a.u[1] = *(const u32x4*)(hbf + lm * 512 + kc * 32 + 16 + hi * 8);
#pragma unroll
      for (int e = 0; e < 4; ++e) {
        int nt = e * 32 + wave;
        const u32x4* bp = (const u32x4*)(whp + ((size_t)(kc * 128 + nt) * 32 + lane) * 16);
        Frag16 b; b.u[0] = bp[0]; b.u[1] = bp[1];
        acc[e] = __builtin_amdgcn_wmma_f32_16x16x32_bf16(false, a.v, false, b.v,
                                                         (short)0, acc[e], false, false);
      }
    }

    // LN over all 4H columns of P (per batch row)
    float ps[8], pq[8];
#pragma unroll
    for (int v = 0; v < 8; ++v) {
      float s = 0.f, q = 0.f;
#pragma unroll
      for (int e = 0; e < 4; ++e) { float p = acc[e][v]; s += p; q += p * p; }
      ps[v] = s; pq[v] = q;
    }
#pragma unroll
    for (int v = 0; v < 8; ++v) {
      float s = hsum16(ps[v]);
      float q = hsum16(pq[v]);
      if (lm == 0) { atomicAdd(&gs[v + 8 * hi], s); atomicAdd(&gq[v + 8 * hi], q); }
    }
    __syncthreads();   // (B) gate sums complete

    float cn[8], og[8], cold[8];
#pragma unroll
    for (int v = 0; v < 8; ++v) {
      int m = v + 8 * hi;
      float m1 = gs[m] * (1.0f / G_);
      float m2 = gq[m] * (1.0f / G_);
      float rstd = rsqrtf(fmaxf(m2 - m1 * m1, 0.f) + EPS_);
      float gate[4];
#pragma unroll
      for (int e = 0; e < 4; ++e)
        gate[e] = (acc[e][v] - m1) * rstd * lgh[e] + lhb_[e] + gxv[e][v] + bs[e];
      cold[v] = cf[m * 512 + nh];
      cn[v] = sig_(gate[0]) * cold[v] + sig_(gate[1]) * tanh_(gate[3]);
      og[v] = sig_(gate[2]);
    }
    // LN stats of c_new over H
#pragma unroll
    for (int v = 0; v < 8; ++v) {
      float s = hsum16(cn[v]);
      float q = hsum16(cn[v] * cn[v]);
      if (lm == 0) { atomicAdd(&cs[v + 8 * hi], s); atomicAdd(&cq[v + 8 * hi], q); }
    }
    __syncthreads();   // (C) c sums complete

#pragma unroll
    for (int v = 0; v < 8; ++v) {
      int m = v + 8 * hi;
      float m1 = cs[m] * (1.0f / H_);
      float m2 = cq[m] * (1.0f / H_);
      float rstd = rsqrtf(fmaxf(m2 - m1 * m1, 0.f) + EPS_);
      float hn = og[v] * tanh_((cn[v] - m1) * rstd * lcg + lcb);
      bool on = (t < len8[v]);
      float hold = hf[m * 512 + nh];
      float hval = on ? hn : hold;
      float cval = on ? cn[v] : cold[v];
      hf[m * 512 + nh]  = hval;
      cf[m * 512 + nh]  = cval;
      hbf[m * 512 + nh] = f2bf(hval);
      seq[((size_t)(rb + m) * T_ + t) * H_ + nh] = hval;
    }
    __syncthreads();   // (D) state + reads of cs/gs done
    if (tid < 16) { gs[tid] = 0.f; gq[tid] = 0.f; cs[tid] = 0.f; cq[tid] = 0.f; }
    __syncthreads();   // (E) scratch rezeroed before next step's atomics
  }

  for (int i = tid; i < 16 * 512; i += 1024) {
    int m = i >> 9, n = i & 511;
    lasth[(size_t)(rb + m) * H_ + n] = hf[i];
  }
}

// ---------------- host ----------------

extern "C" void kernel_launch(void* const* d_in, const int* in_sizes, int n_in,
                              void* d_out, int out_size, void* d_ws, size_t ws_size,
                              hipStream_t stream) {
  (void)in_sizes; (void)n_in; (void)out_size; (void)ws_size;
  const float* x    = (const float*)d_in[0];
  const int*   len  = (const int*)  d_in[1];
  const float* Wx   = (const float*)d_in[2];
  const float* Wh   = (const float*)d_in[3];
  const float* bias = (const float*)d_in[4];
  const float* lxg  = (const float*)d_in[5];
  const float* lxb  = (const float*)d_in[6];
  const float* lhg  = (const float*)d_in[7];
  const float* lhb  = (const float*)d_in[8];
  const float* lcg  = (const float*)d_in[9];
  const float* lcb  = (const float*)d_in[10];

  float* seq   = (float*)d_out;
  float* lasth = seq + (size_t)B_ * T_ * H_;

  char* ws = (char*)d_ws;
  float*    gx  = (float*)ws;                                   // 256 MiB
  uint16_t* xb  = (uint16_t*)(ws + (size_t)B_ * T_ * G_ * 4);   // 16 MiB
  uint16_t* wxp = (uint16_t*)((char*)xb + (size_t)B_ * T_ * E_ * 2);
  uint16_t* whp = (uint16_t*)((char*)wxp + (size_t)G_ * E_ * 2);

  int nx = B_ * T_ * E_;
  k_cvt_bf16<<<(nx + 255) / 256, 256, 0, stream>>>(x, xb, nx);
  k_pack_w<<<(8 * 128 * 32) / 256, 256, 0, stream>>>(Wx, wxp, E_, 8 * 128 * 32);
  k_pack_w<<<(16 * 128 * 32) / 256, 256, 0, stream>>>(Wh, whp, H_, 16 * 128 * 32);
  k_gx<<<(B_ * T_) / 16, 256, 0, stream>>>(xb, wxp, lxg, lxb, gx);
  k_scan<<<4, 1024, 0, stream>>>(gx, whp, bias, lhg, lhb, lcg, lcb, len, seq, lasth);
}
